// InputFeedRNNDecoder_18124761989796
// MI455X (gfx1250) — matrix operand order repaired and checked by reading, not compiled, not run on hardware
//
#include <hip/hip_runtime.h>

// ---------------------------------------------------------------------------
// CDNA5 feature detection: async global->LDS copies (ASYNCcnt-tracked)
// ---------------------------------------------------------------------------
#if defined(__has_builtin)
#if __has_builtin(__builtin_amdgcn_global_load_async_to_lds_b128) && \
    __has_builtin(__builtin_amdgcn_s_wait_asynccnt)
#define USE_ASYNC_COPY 1
#endif
#endif
#ifndef USE_ASYNC_COPY
#define USE_ASYNC_COPY 0
#endif

// The builtin's parameters are generic pointers to 4-int vectors
// (per hipcc diagnostic: 'int __attribute__((__vector_size__(4*sizeof(int)))) *').
typedef int vint4_t __attribute__((vector_size(16)));

// ---------------------------------------------------------------------------
// Types / helpers
// ---------------------------------------------------------------------------
typedef __attribute__((ext_vector_type(16))) __bf16 v16bf;
typedef __attribute__((ext_vector_type(8)))  float  v8f;

union FragB16 {
  v16bf v;
  uint4 q[2];
};

constexpr int BM = 128;        // block tile M
constexpr int BN = 128;        // block tile N
constexpr int BK = 32;         // K slab
constexpr int SSTR = BK + 8;   // LDS row stride in bf16 elems (80B, bank-spread)
constexpr int NTHREADS = 256;  // 8 waves

__device__ __forceinline__ unsigned short f2bf(float f) {
  // round-to-nearest-even fp32 -> bf16
  unsigned int u = __float_as_uint(f);
  u += 0x7FFFu + ((u >> 16) & 1u);
  return (unsigned short)(u >> 16);
}

__device__ __forceinline__ float sigmoidf_(float x) {
  return 1.0f / (1.0f + __expf(-x));
}

// A fragment (16x32 bf16) from LDS slab (row-major, stride SSTR).
// lanes 0-15: rows M0..15, K {0..7, 16..23}; lanes 16-31: K {8..15, 24..31}.
__device__ __forceinline__ v16bf lds_a_frag(const unsigned short* __restrict__ sA,
                                            int m0, int lane) {
  const int row = m0 + (lane & 15);
  const int kh  = (lane >> 4) << 3;            // 0 or 8
  const unsigned short* p = sA + row * SSTR + kh;
  FragB16 f;
  f.q[0] = *(const uint4*)(p);                 // K k..k+7
  f.q[1] = *(const uint4*)(p + 16);            // K k+16..k+23
  return f.v;
}

// B fragment (32x16 bf16) from LDS slab of W rows (cols of output).
// lanes 0-15: col n, K {0..7, 16..23}; lanes 16-31: col n, K {8..15, 24..31}.
__device__ __forceinline__ v16bf lds_b_frag(const unsigned short* __restrict__ sB,
                                            int n0, int lane) {
  const int col = n0 + (lane & 15);
  const int kb  = (lane >> 4) << 3;            // 0 or 8
  const unsigned short* p = sB + col * SSTR + kb;
  FragB16 f;
  f.q[0] = *(const uint4*)(p);
  f.q[1] = *(const uint4*)(p + 16);            // K k+16..k+23
  return f.v;
}

// ---------------------------------------------------------------------------
// Dual-source WMMA GEMM with LDS-staged, double-buffered K slabs:
//   C[M x N] = A0[M x K0] * W0[N x K0]^T + A1[M x K1] * W1[N x K1]^T + biases
// 256 threads = 8 waves (4M x 2N), wave tile 32M x 64N, block tile 128 x 128.
// ---------------------------------------------------------------------------
__global__ __launch_bounds__(256)
void gemm_dual_wmma(const unsigned short* __restrict__ A0,
                    const unsigned short* __restrict__ W0, int K0,
                    const unsigned short* __restrict__ A1,
                    const unsigned short* __restrict__ W1, int K1,
                    const float* __restrict__ bias0,
                    const float* __restrict__ bias1,
                    float* __restrict__ C, float* __restrict__ Cdup, int ldc,
                    unsigned short* __restrict__ Cb16,
                    unsigned short* __restrict__ Cb16dup, int ldcb) {
  __shared__ alignas(16) unsigned short As[2][BM * SSTR];
  __shared__ alignas(16) unsigned short Bs[2][BN * SSTR];

  const int tid  = threadIdx.x;
  const int lane = tid & 31;
  const int wv   = tid >> 5;       // 0..7
  const int wm   = wv >> 1;        // 0..3 -> M
  const int wn   = wv & 1;         // 0..1 -> N
  const int mBase = blockIdx.y * BM;
  const int nBase = blockIdx.x * BN;
  const int m0 = wm * 32;          // within slab
  const int n0 = wn * 64;

  const int nk0 = K0 >> 5;
  const int nk1 = (A1 != nullptr) ? (K1 >> 5) : 0;
  const int nk  = nk0 + nk1;

  const v8f vzero = {0.f, 0.f, 0.f, 0.f, 0.f, 0.f, 0.f, 0.f};
  v8f acc[2][4];
#pragma unroll
  for (int mt = 0; mt < 2; ++mt)
#pragma unroll
    for (int nt = 0; nt < 4; ++nt) acc[mt][nt] = vzero;

  // Cooperative stage of one 128x32 A slab + 128x32 B slab into LDS buffer.
  // 512 16-byte chunks per slab; each of 256 threads moves 2 chunks per slab.
  auto stage = [&](int buf, int step) {
    const unsigned short* gA;
    const unsigned short* gW;
    int ld, k0;
    if (step < nk0) { gA = A0; gW = W0; ld = K0; k0 = step << 5; }
    else            { gA = A1; gW = W1; ld = K1; k0 = (step - nk0) << 5; }
#pragma unroll
    for (int c = 0; c < 2; ++c) {
      const int chunk = tid + c * NTHREADS;              // 0..511
      const int row = chunk >> 2;                        // 0..127
      const int kc  = (chunk & 3) << 3;                  // 0,8,16,24 elems
      const uint4* ga = (const uint4*)(gA + (size_t)(mBase + row) * ld + k0 + kc);
      const uint4* gw = (const uint4*)(gW + (size_t)(nBase + row) * ld + k0 + kc);
      unsigned short* la = &As[buf][row * SSTR + kc];
      unsigned short* lb = &Bs[buf][row * SSTR + kc];
#if USE_ASYNC_COPY
      __builtin_amdgcn_global_load_async_to_lds_b128(
          (vint4_t*)ga, (vint4_t*)la, 0, 0);
      __builtin_amdgcn_global_load_async_to_lds_b128(
          (vint4_t*)gw, (vint4_t*)lb, 0, 0);
#else
      *(uint4*)la = *ga;
      *(uint4*)lb = *gw;
#endif
    }
  };

  stage(0, 0);
  for (int i = 0; i < nk; ++i) {
    const int buf = i & 1;
    const bool hasNext = (i + 1) < nk;
    if (hasNext) stage(buf ^ 1, i + 1);        // overlap next slab with compute
#if USE_ASYNC_COPY
    if (hasNext) __builtin_amdgcn_s_wait_asynccnt(4);  // keep next 4 in flight
    else         __builtin_amdgcn_s_wait_asynccnt(0);
#endif
    __syncthreads();                           // current slab visible to all

    const unsigned short* sA = As[buf];
    const unsigned short* sB = Bs[buf];
    const v16bf a0 = lds_a_frag(sA, m0,      lane);
    const v16bf a1 = lds_a_frag(sA, m0 + 16, lane);
#pragma unroll
    for (int nt = 0; nt < 4; ++nt) {
      const v16bf b = lds_b_frag(sB, n0 + nt * 16, lane);
      acc[0][nt] = __builtin_amdgcn_wmma_f32_16x16x32_bf16(
          false, a0, false, b, (short)0, acc[0][nt], false, false);
      acc[1][nt] = __builtin_amdgcn_wmma_f32_16x16x32_bf16(
          false, a1, false, b, (short)0, acc[1][nt], false, false);
    }
    __syncthreads();                           // done reading before overwrite
  }

  // Epilogue. C element r of lane l: row = m0 + mt*16 + r + 8*(l>=16),
  // col = n0 + nt*16 + (l&15).
  const int rbase = (lane >> 4) << 3;
#pragma unroll
  for (int nt = 0; nt < 4; ++nt) {
    const int col = nBase + n0 + nt * 16 + (lane & 15);
    float bb = 0.0f;
    if (bias0) bb += bias0[col];
    if (bias1) bb += bias1[col];
#pragma unroll
    for (int mt = 0; mt < 2; ++mt) {
#pragma unroll
      for (int r = 0; r < 8; ++r) {
        const int row = mBase + m0 + mt * 16 + rbase + r;
        const float val = acc[mt][nt][r] + bb;
        C[(size_t)row * ldc + col] = val;
        if (Cdup) Cdup[(size_t)row * ldc + col] = val;
        if (Cb16) {
          const unsigned short hv = f2bf(val);
          Cb16[(size_t)row * ldcb + col] = hv;
          if (Cb16dup) Cb16dup[(size_t)row * ldcb + col] = hv;
        }
      }
    }
  }
}

// ---------------------------------------------------------------------------
// fp32 -> bf16 cast, 4 elements/thread
// ---------------------------------------------------------------------------
__global__ void cast_f32_bf16_x4(const float4* __restrict__ s,
                                 ushort4* __restrict__ d, int n4) {
  const int i = blockIdx.x * blockDim.x + threadIdx.x;
  if (i >= n4) return;
  const float4 v = s[i];
  ushort4 o;
  o.x = f2bf(v.x); o.y = f2bf(v.y); o.z = f2bf(v.z); o.w = f2bf(v.w);
  d[i] = o;
}

// ---------------------------------------------------------------------------
// Embedding gather + bf16 cast: x[b, :] = bf16(table[tok[b], :]); E = 2048
// ---------------------------------------------------------------------------
__global__ void embed_gather_bf16(const int* __restrict__ tok,
                                  const float4* __restrict__ table,
                                  ushort4* __restrict__ x) {
  const int i  = blockIdx.x * blockDim.x + threadIdx.x;  // 0 .. B*512-1
  const int b  = i >> 9;                                 // 512 float4 per row
  const int e4 = i & 511;
  const int row = tok[b];
  const float4 v = table[(size_t)row * 512 + e4];
  ushort4 o;
  o.x = f2bf(v.x); o.y = f2bf(v.y); o.z = f2bf(v.z); o.w = f2bf(v.w);
  x[i] = o;
}

// ---------------------------------------------------------------------------
// LSTM pointwise: gates [B,4096] (i,f,g,o), c0 [B,1024] -> h
// ---------------------------------------------------------------------------
__global__ void lstm_pointwise(const float* __restrict__ gates,
                               const float* __restrict__ c0,
                               float* __restrict__ hA,
                               float* __restrict__ hB,            // optional
                               unsigned short* __restrict__ hb16) // optional
{
  const int idx = blockIdx.x * blockDim.x + threadIdx.x;  // 0 .. B*1024-1
  const int b = idx >> 10;
  const int j = idx & 1023;
  const float* g = gates + (size_t)b * 4096;
  const float ig = sigmoidf_(g[j]);
  const float fg = sigmoidf_(g[1024 + j]);
  const float gg = tanhf(g[2048 + j]);
  const float og = sigmoidf_(g[3072 + j]);
  const float cn = fg * c0[idx] + ig * gg;
  const float h  = og * tanhf(cn);
  hA[idx] = h;
  if (hB)   hB[idx] = h;
  if (hb16) hb16[idx] = f2bf(h);
}

// ---------------------------------------------------------------------------
// Launch
// ---------------------------------------------------------------------------
extern "C" void kernel_launch(void* const* d_in, const int* in_sizes, int n_in,
                              void* d_out, int out_size, void* d_ws, size_t ws_size,
                              hipStream_t stream) {
  (void)in_sizes; (void)n_in; (void)out_size; (void)ws_size;

  const int* tok          = (const int*)   d_in[0];
  const float* enc        = (const float*) d_in[1];   // [4096, 1024]
  const float* bridge_W   = (const float*) d_in[2];   // [512, 1024]
  const float* bridge_b   = (const float*) d_in[3];   // [512]
  const float* emb_table  = (const float*) d_in[4];   // [32000, 2048]
  const float* w_ih1      = (const float*) d_in[5];   // [4096, 2048]
  const float* w_hh1      = (const float*) d_in[6];   // [4096, 1024]
  const float* b_ih1      = (const float*) d_in[7];
  const float* b_hh1      = (const float*) d_in[8];
  const float* w_ih2      = (const float*) d_in[9];   // [4096, 1024]
  const float* w_hh2      = (const float*) d_in[10];  // [4096, 1024]
  const float* b_ih2      = (const float*) d_in[11];
  const float* b_hh2      = (const float*) d_in[12];
  float* out              = (float*) d_out;           // [3 * 4096 * 1024]

  // workspace layout (byte offsets, all 256B-aligned)
  unsigned char* base = (unsigned char*)d_ws;
  unsigned short* w_ih1_b   = (unsigned short*)(base + 0);          // 16 MiB
  unsigned short* w_hh1_b   = (unsigned short*)(base + 16777216);   //  8 MiB
  unsigned short* w_ih2_b   = (unsigned short*)(base + 25165824);   //  8 MiB
  unsigned short* w_hh2_b   = (unsigned short*)(base + 33554432);   //  8 MiB
  unsigned short* bridgeW_b = (unsigned short*)(base + 41943040);   //  1 MiB
  unsigned short* enc_b     = (unsigned short*)(base + 42991616);   //  8 MiB
  unsigned short* x_b       = (unsigned short*)(base + 51380224);   // 16 MiB
  float*          hc0_f     = (float*)         (base + 68157440);   // 16 MiB
  unsigned short* hc0_b     = (unsigned short*)(base + 84934656);   //  8 MiB
  unsigned short* h1_b      = (unsigned short*)(base + 93323264);   //  8 MiB
  float*          gates     = (float*)         (base + 101711872);  // 64 MiB

  const int T = 256;

  // 1) bf16 casts of weights + encoder state
  cast_f32_bf16_x4<<<(4096*2048/4 + T - 1)/T, T, 0, stream>>>(
      (const float4*)w_ih1, (ushort4*)w_ih1_b, 4096*2048/4);
  cast_f32_bf16_x4<<<(4096*1024/4 + T - 1)/T, T, 0, stream>>>(
      (const float4*)w_hh1, (ushort4*)w_hh1_b, 4096*1024/4);
  cast_f32_bf16_x4<<<(4096*1024/4 + T - 1)/T, T, 0, stream>>>(
      (const float4*)w_ih2, (ushort4*)w_ih2_b, 4096*1024/4);
  cast_f32_bf16_x4<<<(4096*1024/4 + T - 1)/T, T, 0, stream>>>(
      (const float4*)w_hh2, (ushort4*)w_hh2_b, 4096*1024/4);
  cast_f32_bf16_x4<<<(512*1024/4 + T - 1)/T, T, 0, stream>>>(
      (const float4*)bridge_W, (ushort4*)bridgeW_b, 512*1024/4);
  cast_f32_bf16_x4<<<(4096*1024/4 + T - 1)/T, T, 0, stream>>>(
      (const float4*)enc, (ushort4*)enc_b, 4096*1024/4);

  // 2) embedding gather -> x bf16 [4096, 2048]
  embed_gather_bf16<<<(4096*512)/T, T, 0, stream>>>(
      tok, (const float4*)emb_table, (ushort4*)x_b);

  // 3) bridge GEMM: h0 = enc @ bridge_W^T + b ; hc0 = [h0, h0]
  //    M=4096, N=512, K=1024 ; grid (512/128, 4096/128)
  gemm_dual_wmma<<<dim3(4, 32), NTHREADS, 0, stream>>>(
      enc_b, bridgeW_b, 1024,
      nullptr, nullptr, 0,
      bridge_b, nullptr,
      hc0_f, hc0_f + 512, 1024,
      hc0_b, hc0_b + 512, 1024);

  // 4) layer-1 gates: x @ w_ih1^T + hc0 @ w_hh1^T + b_ih1 + b_hh1
  //    M=4096, N=4096 ; grid (4096/128, 4096/128)
  gemm_dual_wmma<<<dim3(32, 32), NTHREADS, 0, stream>>>(
      x_b, w_ih1_b, 2048,
      hc0_b, w_hh1_b, 1024,
      b_ih1, b_hh1,
      gates, nullptr, 4096,
      nullptr, nullptr, 0);

  // 5) pointwise LSTM 1 -> h1 (hidden_state[0]) + bf16 copy for layer 2
  lstm_pointwise<<<(4096*1024)/T, T, 0, stream>>>(
      gates, hc0_f, out + (size_t)4096*1024, nullptr, h1_b);

  // 6) layer-2 gates: h1 @ w_ih2^T + hc0 @ w_hh2^T + b_ih2 + b_hh2
  gemm_dual_wmma<<<dim3(32, 32), NTHREADS, 0, stream>>>(
      h1_b, w_ih2_b, 1024,
      hc0_b, w_hh2_b, 1024,
      b_ih2, b_hh2,
      gates, nullptr, 4096,
      nullptr, nullptr, 0);

  // 7) pointwise LSTM 2 -> h2 = output and hidden_state[1]
  lstm_pointwise<<<(4096*1024)/T, T, 0, stream>>>(
      gates, hc0_f, out, out + (size_t)2*4096*1024, nullptr);
}
